// EdgeAttention_29283087024206
// MI455X (gfx1250) — compile-verified, matrix operand order checked by reading, not run on hardware
//
#include <hip/hip_runtime.h>
#include <hip/hip_bf16.h>

// ---------------------------------------------------------------------------
// EdgeAttention on MI455X (gfx1250, wave32, WMMA)
//   x:[8192,256] f32, adj:[8192,8192] i32, W:[256,256] f32
//   h = x@W ; e = h h^T ; out = softmax(where(adj>0, e, -9e15), axis=1)
// Strategy:
//   K1: h = x@W in f32, stored to d_ws as fp16 (4 MB -> lives in 192MB L2)
//   K2: e tiles via v_wmma_f32_16x16x32_f16, 64x64 per wave, fused adj mask,
//       masked scores written once to d_out (256 MB)
//   K3: in-place row softmax on d_out
// HBM floor: adj(256MB)+out write(256MB)+softmax rw(512MB) ~= 45us @23.3TB/s
// ---------------------------------------------------------------------------

typedef __attribute__((ext_vector_type(16))) _Float16 v16h;
typedef __attribute__((ext_vector_type(8)))  _Float16 v8h;
typedef __attribute__((ext_vector_type(8)))  float    v8f;

constexpr int kN = 8192;   // nodes
constexpr int kF = 256;    // features
constexpr float kNegInf = -9.0e15f;

// ---------------- Kernel 1: h = x @ W, write fp16 -------------------------
__global__ __launch_bounds__(256) void gcn_h_kernel(const float* __restrict__ x,
                                                    const float* __restrict__ w,
                                                    _Float16* __restrict__ h) {
  __shared__ float xs[kF];
  const int n = blockIdx.x;
  const int f = threadIdx.x;
  xs[f] = x[(size_t)n * kF + f];
  __syncthreads();
  float acc = 0.0f;
#pragma unroll 8
  for (int k = 0; k < kF; ++k)
    acc = fmaf(xs[k], w[(size_t)k * kF + f], acc);  // coalesced across f
  h[(size_t)n * kF + f] = (_Float16)acc;
}

// ---------------- Kernel 2: masked scores via WMMA ------------------------
// Per wave: 64x64 output block = 4x4 tiles of 16x16, K-loop in steps of 32.
// A and B fragments are both rows of h (e = h h^T), loaded per the 16-bit
// fragment striping: lane l (l<16) row=l holds K {0..7, 16..23}; lane l+16
// holds row l's K {8..15, 24..31} -> two b128 loads per fragment per lane.
__global__ __launch_bounds__(128) void gcn_score_kernel(const _Float16* __restrict__ h,
                                                        const int* __restrict__ adj,
                                                        float* __restrict__ out) {
  const int lane = threadIdx.x & 31;
  const int wave = threadIdx.x >> 5;           // 0..3
  const int row0 = blockIdx.y * 128 + (wave >> 1) * 64;
  const int col0 = blockIdx.x * 128 + (wave & 1) * 64;
  const int lrow = lane & 15;
  const int sel  = lane >> 4;                  // 0: K low-8 half, 1: high-8

  v8f acc[4][4] = {};                          // 128 VGPRs of f32 accumulators

  for (int k0 = 0; k0 < kF; k0 += 32) {
    v16h a[4], b[4];
#pragma unroll
    for (int mt = 0; mt < 4; ++mt) {
      const _Float16* p = h + (size_t)(row0 + mt * 16 + lrow) * kF + k0 + sel * 8;
      union { v16h v; v8h half8[2]; } u;
      u.half8[0] = *(const v8h*)p;             // K = k0 + sel*8 .. +7
      u.half8[1] = *(const v8h*)(p + 16);      // K = k0+16 + sel*8 .. +7
      a[mt] = u.v;
    }
#pragma unroll
    for (int nt = 0; nt < 4; ++nt) {
      const _Float16* p = h + (size_t)(col0 + nt * 16 + lrow) * kF + k0 + sel * 8;
      union { v16h v; v8h half8[2]; } u;
      u.half8[0] = *(const v8h*)p;
      u.half8[1] = *(const v8h*)(p + 16);
      b[nt] = u.v;
    }
#pragma unroll
    for (int mt = 0; mt < 4; ++mt)
#pragma unroll
      for (int nt = 0; nt < 4; ++nt)
        acc[mt][nt] = __builtin_amdgcn_wmma_f32_16x16x32_f16(
            /*neg_a=*/false, a[mt], /*neg_b=*/false, b[nt],
            /*c_mod=*/(short)0, acc[mt][nt],
            /*reuse_a=*/false, /*reuse_b=*/false);
  }

  // C/D layout: lane l, VGPR v -> row M = v + sel*8, col N = l&15.
  // Fuse adj mask; single pass over the 8192x8192 plane.
#pragma unroll
  for (int mt = 0; mt < 4; ++mt) {
#pragma unroll
    for (int v = 0; v < 8; ++v) {
      const int gr = row0 + mt * 16 + sel * 8 + v;
      const size_t rbase = (size_t)gr * kN;
#pragma unroll
      for (int nt = 0; nt < 4; ++nt) {
        const int gc = col0 + nt * 16 + lrow;
        const int m = adj[rbase + gc];
        out[rbase + gc] = (m > 0) ? acc[mt][nt][v] : kNegInf;
      }
    }
  }
}

// ---------------- Kernel 3: row softmax in place --------------------------
__global__ __launch_bounds__(256) void gcn_softmax_kernel(float* __restrict__ out) {
  __shared__ float red[256];
  const int tid = threadIdx.x;
  float* r = out + (size_t)blockIdx.x * kN;

  float m = -3.0e38f;
  for (int c = tid; c < kN; c += 256) m = fmaxf(m, r[c]);
  red[tid] = m;
  __syncthreads();
  for (int s = 128; s > 0; s >>= 1) {
    if (tid < s) red[tid] = fmaxf(red[tid], red[tid + s]);
    __syncthreads();
  }
  m = red[0];
  __syncthreads();

  float sum = 0.0f;
  for (int c = tid; c < kN; c += 256) sum += __expf(r[c] - m);
  red[tid] = sum;
  __syncthreads();
  for (int s = 128; s > 0; s >>= 1) {
    if (tid < s) red[tid] += red[tid + s];
    __syncthreads();
  }
  const float inv = 1.0f / red[0];

  for (int c = tid; c < kN; c += 256) r[c] = __expf(r[c] - m) * inv;
}

// ---------------------------------------------------------------------------
extern "C" void kernel_launch(void* const* d_in, const int* in_sizes, int n_in,
                              void* d_out, int out_size, void* d_ws, size_t ws_size,
                              hipStream_t stream) {
  const float* x   = (const float*)d_in[0];   // [8192,256] f32
  const int*   adj = (const int*)d_in[1];     // [8192,8192] i32
  const float* w   = (const float*)d_in[2];   // [256,256] f32
  float* out = (float*)d_out;                 // [8192,8192] f32
  _Float16* h = (_Float16*)d_ws;              // 8192*256 fp16 = 4 MB scratch

  gcn_h_kernel<<<kN, 256, 0, stream>>>(x, w, h);
  gcn_score_kernel<<<dim3(kN / 128, kN / 128), 128, 0, stream>>>(h, adj, out);
  gcn_softmax_kernel<<<kN, 256, 0, stream>>>(out);
}
